// LinearQwen3VLMoeTextSparseMoeBlock_78554951844231
// MI455X (gfx1250) — matrix operand order repaired and checked by reading, not compiled, not run on hardware
//
#include <hip/hip_runtime.h>

// ---------------- problem constants (from reference) ----------------
#define HDIM 2048   // hidden size H
#define IDIM 768    // intermediate size I
#define NEXP 32     // experts E
#define TOPK 8
#define TKC  64     // K-tile per barrier (2 x WMMA K=32 sub-chunks)
#define BN   64     // N columns per block (4 waves x 16)
#define BM   16     // M rows (tokens) per block

typedef __attribute__((ext_vector_type(16))) __bf16       v16bf;
typedef __attribute__((ext_vector_type(8)))  float        v8f;
typedef __attribute__((ext_vector_type(8)))  unsigned int v8u;

static __device__ __forceinline__ unsigned short f2bf(float f) {
    unsigned u = __float_as_uint(f);
    u += 0x7FFFu + ((u >> 16) & 1u);          // round-to-nearest-even
    return (unsigned short)(u >> 16);
}
static __device__ __forceinline__ unsigned pack2bf(float lo, float hi) {
    return (unsigned)f2bf(lo) | ((unsigned)f2bf(hi) << 16);
}

// ---------------- 1: zero output + counts ----------------
__global__ void init_kernel(float* __restrict__ out, int n, int* __restrict__ counts) {
    int i = blockIdx.x * blockDim.x + threadIdx.x;
    if (i < n) out[i] = 0.0f;
    if (i < NEXP) counts[i] = 0;
}

// ---------------- 2: router (one wave32 per token; lane == expert) ----------------
__global__ __launch_bounds__(128) void router_kernel(
    const float* __restrict__ x, const float* __restrict__ gate_w,
    float* __restrict__ topk_w, int* __restrict__ topk_idx,
    int* __restrict__ counts, int T)
{
    int t    = blockIdx.x * 4 + (threadIdx.x >> 5);
    int lane = threadIdx.x & 31;
    if (t >= T) return;

    const float4* xp = (const float4*)(x + (long)t * HDIM);
    const float4* gp = (const float4*)(gate_w + (long)lane * HDIM);
    float acc = 0.0f;
    for (int i = 0; i < HDIM / 4; ++i) {
        float4 a = xp[i], b = gp[i];
        acc += a.x * b.x + a.y * b.y + a.z * b.z + a.w * b.w;
    }
    // wave-wide softmax (fp32)
    float m = acc;
    for (int s = 16; s; s >>= 1) m = fmaxf(m, __shfl_xor(m, s, 32));
    float p = __expf(acc - m);
    float sum = p;
    for (int s = 16; s; s >>= 1) sum += __shfl_xor(sum, s, 32);
    float prob = p / sum;

    // iterative top-8 argmax; deterministic lane tie-break in low 5 bits
    float cur = prob, wsel = 0.0f, ssum = 0.0f;
    int isel = 0;
    #pragma unroll
    for (int k = 0; k < TOPK; ++k) {
        unsigned key = (__float_as_uint(cur) & 0xFFFFFFE0u) | (unsigned)lane;
        for (int s = 16; s; s >>= 1) {
            unsigned o = __shfl_xor(key, s, 32);
            key = (o > key) ? o : key;
        }
        int win = (int)(key & 31u);
        float wv = __shfl(prob, win, 32);
        ssum += wv;
        if (lane == win) cur = 0.0f;
        if (lane == k) { wsel = wv; isel = win; }
    }
    if (lane < TOPK) {
        topk_w  [t * TOPK + lane] = wsel / ssum;
        topk_idx[t * TOPK + lane] = isel;
        atomicAdd(&counts[isel], 1);
    }
}

// ---------------- 3: serial scan (32 experts, pad segments to 16) ----------------
__global__ void scan_kernel(const int* __restrict__ counts,
                            int* __restrict__ offsets, int* __restrict__ cursor)
{
    if (threadIdx.x == 0) {
        int o = 0;
        for (int e = 0; e < NEXP; ++e) {
            offsets[e] = o;
            cursor[e]  = 0;
            o += (counts[e] + 15) & ~15;
        }
        offsets[NEXP] = o;   // total padded rows
    }
}

// ---------------- 4a: scatter (token, weight) into expert segments ----------------
__global__ void scatter_kernel(const int* __restrict__ topk_idx,
                               const float* __restrict__ topk_w,
                               const int* __restrict__ offsets, int* __restrict__ cursor,
                               int* __restrict__ rows_tok, float* __restrict__ rows_w, int n)
{
    int i = blockIdx.x * blockDim.x + threadIdx.x;
    if (i >= n) return;
    int t = i >> 3;
    int e = topk_idx[i];
    int p = atomicAdd(&cursor[e], 1);
    int row = offsets[e] + p;
    rows_tok[row] = t;
    rows_w  [row] = topk_w[i];
}

// ---------------- 4b: expert id per row; zero the pad rows ----------------
__global__ void fill_kernel(const int* __restrict__ offsets, const int* __restrict__ counts,
                            int* __restrict__ rows_tok, float* __restrict__ rows_w,
                            int* __restrict__ rows_exp, int maxRows)
{
    int r = blockIdx.x * blockDim.x + threadIdx.x;
    if (r >= maxRows) return;
    int total = offsets[NEXP];
    if (r >= total) return;
    int e = 0;
    for (int k = 0; k < NEXP; ++k)
        if (r >= offsets[k]) e = k;
    rows_exp[r] = e;
    if (r >= offsets[e] + counts[e]) {   // padding row
        rows_tok[r] = 0;
        rows_w [r]  = 0.0f;
    }
}

// ---- WMMA fragment gather from LDS (wave32 bf16 layouts, ISA 7.12.2) ----
// A 16x32: lane m=lane&15, half=lane>>4; VGPR v holds K = 16*(v>>2)+8*half+2*(v&3), +1
// B 32x16 (stored [n][k] in LDS): VGPR v holds K = 16*(v>>2)+8*half+2*(v&3), +1
static __device__ __forceinline__ v16bf frag_a(const unsigned int* a_lds, int mn, int half, int koff) {
    v8u au;
    #pragma unroll
    for (int v = 0; v < 8; ++v)
        au[v] = a_lds[mn * (TKC / 2) + koff + 8 * (v >> 2) + 4 * half + (v & 3)];
    return __builtin_bit_cast(v16bf, au);
}
static __device__ __forceinline__ v16bf frag_b(const unsigned int* b_lds, int nt, int half, int koff) {
    v8u bu;
    #pragma unroll
    for (int v = 0; v < 8; ++v)
        bu[v] = b_lds[nt * (TKC / 2) + koff + 8 * (v >> 2) + 4 * half + (v & 3)];
    return __builtin_bit_cast(v16bf, bu);
}

// ---------------- 5: gate/up GEMMs + SwiGLU (bf16 WMMA, fp32 acc) ----------------
__global__ __launch_bounds__(128) void moe_gu_kernel(
    const float* __restrict__ x,
    const float* __restrict__ w_gate, const float* __restrict__ w_up,
    const int* __restrict__ offsets,
    const int* __restrict__ rows_tok, const float* __restrict__ rows_w,
    unsigned short* __restrict__ h_buf)
{
    const int e = blockIdx.z;
    const int off0 = offsets[e], off1 = offsets[e + 1];
    const int row0 = off0 + blockIdx.y * BM;
    if (row0 >= off1) return;
    const int i0 = blockIdx.x * BN;

    __shared__ unsigned int a_lds [BM * TKC / 2];   // [m][kpair] bf16x2, 2KB
    __shared__ unsigned int bg_lds[BN * TKC / 2];   // [n][kpair], 8KB
    __shared__ unsigned int bu_lds[BN * TKC / 2];   // 8KB
    __shared__ int   tok_lds[BM];
    __shared__ float w_lds  [BM];

    const int tid  = threadIdx.x;
    const int wv   = tid >> 5;
    const int lane = tid & 31;
    const int half = lane >> 4;
    const int mn   = lane & 15;
    const int nt   = wv * 16 + mn;

    if (tid < BM) {
        tok_lds[tid] = rows_tok[row0 + tid];
        w_lds  [tid] = rows_w [row0 + tid];
    }

    v8f accg = {}; v8f accu = {};
    for (int k0 = 0; k0 < HDIM; k0 += TKC) {
        __syncthreads();
        {   // A tile: gathered x rows, fp32 -> bf16 (16x64)
            #pragma unroll
            for (int r = 0; r < 2; ++r) {
                int idx = tid + 128 * r;            // 0..255
                int row = idx >> 4;                 // 0..15
                int kc  = (idx & 15) * 4;           // 0..60
                const float4 xv = *(const float4*)(x + (long)tok_lds[row] * HDIM + k0 + kc);
                a_lds[row * (TKC / 2) + (kc >> 1) + 0] = pack2bf(xv.x, xv.y);
                a_lds[row * (TKC / 2) + (kc >> 1) + 1] = pack2bf(xv.z, xv.w);
            }
        }
        {   // B tiles: w_gate / w_up [H][I] slice (64x64), stored [n][k] bf16
            unsigned short* bgs = (unsigned short*)bg_lds;
            unsigned short* bus = (unsigned short*)bu_lds;
            #pragma unroll
            for (int r = 0; r < 8; ++r) {
                int idx = tid + 128 * r;            // 0..1023
                int kl  = idx >> 4;                 // 0..63
                int i4  = (idx & 15) * 4;           // 0..60
                long gofs = ((long)e * HDIM + k0 + kl) * IDIM + i0 + i4;
                const float4 gv = *(const float4*)(w_gate + gofs);
                const float4 uv = *(const float4*)(w_up   + gofs);
                bgs[(i4 + 0) * TKC + kl] = f2bf(gv.x);
                bgs[(i4 + 1) * TKC + kl] = f2bf(gv.y);
                bgs[(i4 + 2) * TKC + kl] = f2bf(gv.z);
                bgs[(i4 + 3) * TKC + kl] = f2bf(gv.w);
                bus[(i4 + 0) * TKC + kl] = f2bf(uv.x);
                bus[(i4 + 1) * TKC + kl] = f2bf(uv.y);
                bus[(i4 + 2) * TKC + kl] = f2bf(uv.z);
                bus[(i4 + 3) * TKC + kl] = f2bf(uv.w);
            }
        }
        __syncthreads();

        #pragma unroll
        for (int c = 0; c < 2; ++c) {               // two K=32 sub-chunks
            v16bf A  = frag_a(a_lds,  mn, half, 16 * c);
            v16bf Bg = frag_b(bg_lds, nt, half, 16 * c);
            v16bf Bu = frag_b(bu_lds, nt, half, 16 * c);
            accg = __builtin_amdgcn_wmma_f32_16x16x32_bf16(false, A, false, Bg, (short)0, accg, false, false);
            accu = __builtin_amdgcn_wmma_f32_16x16x32_bf16(false, A, false, Bu, (short)0, accu, false, false);
        }
    }

    // fused SwiGLU + combine weight, store h as bf16 (fast v_rcp_f32 sigmoid)
    #pragma unroll
    for (int r = 0; r < 8; ++r) {
        int m = r + 8 * half;
        float g = accg[r], u = accu[r];
        float sig = __builtin_amdgcn_rcpf(1.0f + __expf(-g));
        float h = g * sig * u * w_lds[m];
        h_buf[(long)(row0 + m) * IDIM + i0 + nt] = f2bf(h);
    }
}

// ---------------- 6: down-proj GEMM + atomic combine ----------------
__global__ __launch_bounds__(128) void moe_down_kernel(
    const unsigned short* __restrict__ h_buf, const float* __restrict__ w_down,
    const int* __restrict__ offsets, const int* __restrict__ rows_tok,
    const int* __restrict__ rows_exp, float* __restrict__ out)
{
    const int row0 = blockIdx.y * BM;
    if (row0 >= offsets[NEXP]) return;
    const int e  = rows_exp[row0];
    const int h0 = blockIdx.x * BN;

    __shared__ unsigned int a_lds[BM * TKC / 2];    // 2KB
    __shared__ unsigned int b_lds[BN * TKC / 2];    // 8KB
    __shared__ int tok_lds[BM];

    const int tid  = threadIdx.x;
    const int wv   = tid >> 5;
    const int lane = tid & 31;
    const int half = lane >> 4;
    const int mn   = lane & 15;
    const int nt   = wv * 16 + mn;

    if (tid < BM) tok_lds[tid] = rows_tok[row0 + tid];

    v8f acc = {};
    for (int k0 = 0; k0 < IDIM; k0 += TKC) {
        __syncthreads();
        {   // A: h rows already bf16 (16x64) -> dword copies
            #pragma unroll
            for (int r = 0; r < 4; ++r) {
                int idx = tid + 128 * r;            // 0..511 dwords
                int m   = idx >> 5;
                int kd  = idx & 31;
                const unsigned int* hp =
                    (const unsigned int*)(h_buf + (long)(row0 + m) * IDIM + k0);
                a_lds[m * (TKC / 2) + kd] = hp[kd];
            }
        }
        {   // B: w_down [I][H] slice (64x64) fp32 -> bf16, [n][k]
            unsigned short* bs = (unsigned short*)b_lds;
            #pragma unroll
            for (int r = 0; r < 8; ++r) {
                int idx = tid + 128 * r;
                int kl  = idx >> 4;                 // 0..63
                int i4  = (idx & 15) * 4;
                const float4 dv =
                    *(const float4*)(w_down + ((long)e * IDIM + k0 + kl) * HDIM + h0 + i4);
                bs[(i4 + 0) * TKC + kl] = f2bf(dv.x);
                bs[(i4 + 1) * TKC + kl] = f2bf(dv.y);
                bs[(i4 + 2) * TKC + kl] = f2bf(dv.z);
                bs[(i4 + 3) * TKC + kl] = f2bf(dv.w);
            }
        }
        __syncthreads();

        #pragma unroll
        for (int c = 0; c < 2; ++c) {
            v16bf A = frag_a(a_lds, mn, half, 16 * c);
            v16bf B = frag_b(b_lds, nt, half, 16 * c);
            acc = __builtin_amdgcn_wmma_f32_16x16x32_bf16(false, A, false, B, (short)0, acc, false, false);
        }
    }

    #pragma unroll
    for (int r = 0; r < 8; ++r) {
        int m = r + 8 * half;
        atomicAdd(&out[(long)tok_lds[m] * HDIM + h0 + nt], acc[r]);
    }
}

// ---------------- host-side orchestration ----------------
extern "C" void kernel_launch(void* const* d_in, const int* in_sizes, int n_in,
                              void* d_out, int out_size, void* d_ws, size_t ws_size,
                              hipStream_t stream) {
    const float* x      = (const float*)d_in[0];   // [T,H]
    const float* gate_w = (const float*)d_in[1];   // [E,H]
    const float* w_gate = (const float*)d_in[2];   // [E,H,I]
    const float* w_up   = (const float*)d_in[3];   // [E,H,I]
    const float* w_down = (const float*)d_in[4];   // [E,I,H]
    float* out = (float*)d_out;

    const int T = in_sizes[0] / HDIM;                       // 2048
    const int maxRows = ((T * TOPK + NEXP * 16) + 15) & ~15;

    char* ws = (char*)d_ws;
    size_t o = 0;
    float* topk_w   = (float*)(ws + o); o += (size_t)T * TOPK * 4;
    int*   topk_idx = (int*)  (ws + o); o += (size_t)T * TOPK * 4;
    int*   counts   = (int*)  (ws + o); o += 256;
    int*   offsets  = (int*)  (ws + o); o += 256;
    int*   cursor   = (int*)  (ws + o); o += 256;
    int*   rows_tok = (int*)  (ws + o); o += (size_t)maxRows * 4;
    float* rows_w   = (float*)(ws + o); o += (size_t)maxRows * 4;
    int*   rows_exp = (int*)  (ws + o); o += (size_t)maxRows * 4;
    o = (o + 255) & ~(size_t)255;
    unsigned short* h_buf = (unsigned short*)(ws + o);      // [maxRows, I] bf16

    init_kernel   <<<(T * HDIM + 255) / 256, 256, 0, stream>>>(out, T * HDIM, counts);
    router_kernel <<<(T + 3) / 4, 128, 0, stream>>>(x, gate_w, topk_w, topk_idx, counts, T);
    scan_kernel   <<<1, 32, 0, stream>>>(counts, offsets, cursor);
    scatter_kernel<<<(T * TOPK + 255) / 256, 256, 0, stream>>>(topk_idx, topk_w, offsets,
                                                               cursor, rows_tok, rows_w, T * TOPK);
    fill_kernel   <<<(maxRows + 255) / 256, 256, 0, stream>>>(offsets, counts, rows_tok,
                                                              rows_w, rows_exp, maxRows);
    dim3 ggu(IDIM / BN, (T + BM - 1) / BM, NEXP);
    moe_gu_kernel <<<ggu, 128, 0, stream>>>(x, w_gate, w_up, offsets, rows_tok, rows_w, h_buf);
    dim3 gdn(HDIM / BN, maxRows / BM, 1);
    moe_down_kernel<<<gdn, 128, 0, stream>>>(h_buf, w_down, offsets, rows_tok, rows_exp, out);
}
